// QuantizerEncoder_49727131353590
// MI455X (gfx1250) — matrix operand order repaired and checked by reading, not compiled, not run on hardware
//
#include <hip/hip_runtime.h>

typedef __attribute__((ext_vector_type(16))) __bf16 v16bf;
typedef __attribute__((ext_vector_type(8)))  float  v8f;
typedef __attribute__((ext_vector_type(4)))  unsigned int u32x4;

union BfOp { u32x4 u[2]; v16bf v; };

__device__ __forceinline__ unsigned bfhi(float f) {
  union { float f; unsigned u; } x; x.f = f; return x.u >> 16;
}
__device__ __forceinline__ float fb(unsigned u) {
  union { unsigned u; float f; } x; x.u = u; return x.f;
}

// ---------------------------------------------------------------------------
// Phase 1a: Gt[m][d'][d] = sum_c wq[m][c][d] * wk[m][c][d']   (fp32, tiny)
// ---------------------------------------------------------------------------
__global__ void __launch_bounds__(128)
qe_gt_kernel(const float* __restrict__ wq, const float* __restrict__ wk,
             float* __restrict__ Gt) {
  const int m  = blockIdx.x >> 7;
  const int dp = blockIdx.x & 127;
  const int d  = threadIdx.x;
  const float* q = wq + (size_t)m * 16384;
  const float* k = wk + (size_t)m * 16384;
  float acc = 0.f;
#pragma unroll 4
  for (int c = 0; c < 128; ++c) acc = fmaf(q[c * 128 + d], k[c * 128 + dp], acc);
  Gt[(size_t)m * 16384 + (size_t)dp * 128 + d] = acc;
}

// ---------------------------------------------------------------------------
// Phase 1b: E[m][k][d] = sum_d' cb[m][k][d'] * Gt[m][d'][d]; split to bf16
// ---------------------------------------------------------------------------
__global__ void __launch_bounds__(128)
qe_e_kernel(const float* __restrict__ cb, const float* __restrict__ Gt,
            unsigned short* __restrict__ Ehi, unsigned short* __restrict__ Elo) {
  const int m  = blockIdx.x >> 8;
  const int kc = blockIdx.x & 255;
  const int d  = threadIdx.x;
  const float* c = cb + ((size_t)m * 256 + kc) * 128;
  const float* g = Gt + (size_t)m * 16384;
  float acc = 0.f;
#pragma unroll 4
  for (int dp = 0; dp < 128; ++dp) acc = fmaf(c[dp], g[dp * 128 + d], acc);
  unsigned hi = bfhi(acc);
  unsigned lo = bfhi(acc - fb(hi << 16));   // residual (exact subtraction)
  size_t o = ((size_t)m * 256 + kc) * 128 + d;
  Ehi[o] = (unsigned short)hi;
  Elo[o] = (unsigned short)lo;
}

// ---------------------------------------------------------------------------
// Phase 2: per group m, scores = q @ E^T with split-bf16 WMMA, fused argmax.
// grid = 256 blocks (m, 64-tile chunk), 512 threads = 16 waves (4/SIMD on one
// WGP: maximizes latency hiding over the WMMA pipe).
// LDS: E stored chunk-major+padded: chunk u=nt*4+kk -> 16 rows x 80B (64B data
// + 16B pad).  Lane address pos*80 => bank starts 20*pos mod 64: each 16-lane
// half tiles all 64 banks exactly once -> conflict-free ds_load_b128.
// Chunk offsets are pure immediates (kk*1280, +5120/6400 lookahead).
// Layout: E_hi[0,81920) E_lo[81920,163840) qstage 16x8704 -> 303104 B total.
// ---------------------------------------------------------------------------
__global__ void __launch_bounds__(512)
qe_main_kernel(const float* __restrict__ latent,
               const unsigned short* __restrict__ EhiG,
               const unsigned short* __restrict__ EloG,
               float* __restrict__ out) {
  extern __shared__ char smem[];
  const int m     = blockIdx.x & 3;
  const int chunk = blockIdx.x >> 2;

  { // stage E into LDS, permuting to chunk-major padded layout
    const u32x4* sh = (const u32x4*)(EhiG + (size_t)m * 32768);
    const u32x4* sl = (const u32x4*)(EloG + (size_t)m * 32768);
    u32x4* dh = (u32x4*)smem;             // uint4 units: 80 per chunk
    u32x4* dl = (u32x4*)(smem + 81920);
    for (int j = threadIdx.x; j < 4096; j += 512) {
      const int code = j >> 4;            // 0..255
      const int i    = j & 15;            // uint4 within row: d = i*8..i*8+7
      const int ntc  = code >> 4;
      const int c    = code & 15;
      // dest uint4 index: chunk (ntc*4 + i/4) * 80 + c*5 + (i%4)
      const int dst = (ntc * 4 + (i >> 2)) * 80 + c * 5 + (i & 3);
      dh[dst] = sh[j];
      dl[dst] = sl[j];
    }
  }
  __syncthreads();

  const int wave = threadIdx.x >> 5;     // 0..15
  const int lane = threadIdx.x & 31;
  const int pos  = lane & 15;   // spatial position within tile (= M row)
  const int drow = lane >> 4;   // which of the 2 d-rows this lane loads
  char* qbase = smem + 163840 + wave * 8704;
  unsigned short* qh = (unsigned short*)qbase;           // [16][136] bf16 hi
  unsigned short* ql = (unsigned short*)(qbase + 4352);  // [16][136] bf16 lo

  for (int tt = wave; tt < 64; tt += 16) {
    const int t  = chunk * 64 + tt;       // 16-position tile id
    const int n  = t >> 8;
    const int h  = (t >> 2) & 63;
    const int w0 = (t & 3) << 4;
    const float* lp =
        latent + (size_t)(n * 512 + m * 128) * 4096 + h * 64 + w0 + pos;

    // ---- stage q tile [16 pos][128 d] -> LDS as bf16 hi/lo (transposed) ----
#pragma unroll 4
    for (int i = 0; i < 64; ++i) {
      float v  = lp[(size_t)(2 * i + drow) * 4096];  // coalesced 64B per half
      float pv = __shfl_xor(v, 16, 32);
      float fe = drow ? pv : v;                      // d = 2i   value
      float fo = drow ? v : pv;                      // d = 2i+1 value
      unsigned he = bfhi(fe), ho = bfhi(fo);
      if (drow == 0) {
        *(unsigned*)((char*)qh + pos * 272 + 4 * i) = he | (ho << 16);
      } else {
        unsigned le = bfhi(fe - fb(he << 16));
        unsigned lo = bfhi(fo - fb(ho << 16));
        *(unsigned*)((char*)ql + pos * 272 + 4 * i) = le | (lo << 16);
      }
    }
    asm volatile("s_wait_dscnt 0" ::: "memory");  // wave-private staging fence

    // ---- build A operands (16x32 bf16 per K-chunk), hi and lo ----
    BfOp Ah[4], Al[4];
    const char* qrh = (const char*)qh + pos * 272;
    const char* qrl = (const char*)ql + pos * 272;
    const int r0 = (lane < 16) ? 0 : 16;  // per-lane K run start (A layout)
#pragma unroll
    for (int kk = 0; kk < 4; ++kk) {
      Ah[kk].u[0] = *(const u32x4*)(qrh + (kk * 32 + r0) * 2);
      Ah[kk].u[1] = *(const u32x4*)(qrh + (kk * 32 + r0 + 8) * 2);
      Al[kk].u[0] = *(const u32x4*)(qrl + (kk * 32 + r0) * 2);
      Al[kk].u[1] = *(const u32x4*)(qrl + (kk * 32 + r0 + 8) * 2);
    }

    // ---- prefetch next tile's latent rows; overlaps with WMMA phase ----
    if (tt + 16 < 64) {
      const int t2  = t + 16;
      const int n2  = t2 >> 8;
      const int h2  = (t2 >> 2) & 63;
      const int w02 = (t2 & 3) << 4;
      const float* lp2 =
          latent + (size_t)(n2 * 512 + m * 128) * 4096 + h2 * 64 + w02 + pos;
#pragma unroll 4
      for (int i = 0; i < 64; ++i)
        __builtin_prefetch(lp2 + (size_t)(2 * i + drow) * 4096, 0, 3);
    }

    float bv[8];
    int   bi[8];
#pragma unroll
    for (int j = 0; j < 8; ++j) { bv[j] = -__builtin_inff(); bi[j] = 0; }

    // ---- software-pipelined WMMA loop: 4 rotating B buffers, 2 chunks ahead
    const int   boff2 = (lane < 16) ? 0 : 32;  // per-lane K half (B layout)
    const char* bh0 = smem + pos * 80 + boff2;           // E_hi lane base
    const char* bl0 = smem + 81920 + pos * 80 + boff2;   // E_lo lane base

    BfOp Bh[4], Bl[4];
#pragma unroll
    for (int p = 0; p < 2; ++p) {      // prologue: chunks u=0,1
      Bh[p].u[0] = *(const u32x4*)(bh0 + p * 1280);
      Bh[p].u[1] = *(const u32x4*)(bh0 + p * 1280 + 16);
      Bl[p].u[0] = *(const u32x4*)(bl0 + p * 1280);
      Bl[p].u[1] = *(const u32x4*)(bl0 + p * 1280 + 16);
    }

#pragma unroll 1
    for (int nt = 0; nt < 16; ++nt) {       // 16 code tiles = 256 codes
      const char* ph = bh0 + nt * 5120;     // one add per nt, addr regs only
      const char* pl = bl0 + nt * 5120;
      v8f acc = {};
#pragma unroll
      for (int kk = 0; kk < 4; ++kk) {      // K = 128 in 4 chunks of 32
        // loads for chunk u+2 -> buffer (kk+2)&3; offsets are immediates
        // (2560,3840,5120,6400); tail overrun reads valid LDS, never consumed
        const int io = (kk + 2) * 1280;
        Bh[(kk + 2) & 3].u[0] = *(const u32x4*)(ph + io);
        Bh[(kk + 2) & 3].u[1] = *(const u32x4*)(ph + io + 16);
        Bl[(kk + 2) & 3].u[0] = *(const u32x4*)(pl + io);
        Bl[(kk + 2) & 3].u[1] = *(const u32x4*)(pl + io + 16);
        // split-bf16: hi*hi + lo*hi + hi*lo  (~fp32-accurate scores)
        acc = __builtin_amdgcn_wmma_f32_16x16x32_bf16(
            false, Ah[kk].v, false, Bh[kk].v, (short)0, acc, false, false);
        acc = __builtin_amdgcn_wmma_f32_16x16x32_bf16(
            false, Al[kk].v, false, Bh[kk].v, (short)0, acc, false, false);
        acc = __builtin_amdgcn_wmma_f32_16x16x32_bf16(
            false, Ah[kk].v, false, Bl[kk].v, (short)0, acc, false, false);
      }
      const int code = nt * 16 + pos;       // D slot: lane holds code N=pos
#pragma unroll
      for (int j = 0; j < 8; ++j) {
        float s = acc[j];
        if (s > bv[j]) { bv[j] = s; bi[j] = code; }  // strict > keeps first max
      }
    }

    // ---- argmax across the 16 lanes of each half (rows M=j(+8)) ----
#pragma unroll
    for (int msk = 1; msk <= 8; msk <<= 1) {
#pragma unroll
      for (int j = 0; j < 8; ++j) {
        float ov = __shfl_xor(bv[j], msk, 32);
        int   oi = __shfl_xor(bi[j], msk, 32);
        if (ov > bv[j] || (ov == bv[j] && oi < bi[j])) { bv[j] = ov; bi[j] = oi; }
      }
    }
    if (pos == 0) {  // lane 0 writes M=0..7, lane 16 writes M=8..15
      const size_t pbase = (size_t)t * 16 + drow * 8;
#pragma unroll
      for (int j = 0; j < 8; ++j)
        out[(pbase + j) * 4 + m] = (float)bi[j];
    }
  }
}

// ---------------------------------------------------------------------------
extern "C" void kernel_launch(void* const* d_in, const int* in_sizes, int n_in,
                              void* d_out, int out_size, void* d_ws, size_t ws_size,
                              hipStream_t stream) {
  (void)in_sizes; (void)n_in; (void)out_size; (void)ws_size;
  const float* latent   = (const float*)d_in[0];
  const float* codebook = (const float*)d_in[1];
  const float* wq       = (const float*)d_in[2];
  const float* wk       = (const float*)d_in[3];
  float* out = (float*)d_out;
  char*  ws  = (char*)d_ws;

  float*          Gt  = (float*)ws;                     // 256 KB
  unsigned short* Ehi = (unsigned short*)(ws + 262144); // 256 KB
  unsigned short* Elo = (unsigned short*)(ws + 524288); // 256 KB

  qe_gt_kernel<<<dim3(512), dim3(128), 0, stream>>>(wq, wk, Gt);
  qe_e_kernel<<<dim3(1024), dim3(128), 0, stream>>>(codebook, Gt, Ehi, Elo);
  // dynamic LDS: 2*81920 (padded E hi/lo) + 16*8704 (q staging) = 303104 B
  qe_main_kernel<<<dim3(256), dim3(512), 303104, stream>>>(latent, Ehi, Elo, out);
}
